// GoEModel_74199855006302
// MI455X (gfx1250) — matrix operand order repaired and checked by reading, not compiled, not run on hardware
//
#include <hip/hip_runtime.h>
#include <math.h>

// ---------------- model dims ----------------
constexpr int BB = 8, SS = 256, DD = 512, HH = 8, DHH = 64;
constexpr int FFF = 2048, EE = 8, VV = 32000, RHH = 256, EP1 = EE + 1;
constexpr int MAX_PATH = 4, MAX_VISITS = 2;
constexpr float NEGV = -1e9f;

// ---------------- WMMA / TDM types ----------------
typedef __bf16 bf16;
typedef __attribute__((ext_vector_type(16))) __bf16 v16bf;
typedef __attribute__((ext_vector_type(8)))  float  v8f;
typedef __attribute__((ext_vector_type(4)))  unsigned int v4u;
typedef __attribute__((ext_vector_type(8)))  int    v8i;
typedef __attribute__((ext_vector_type(4)))  int    v4i;

__device__ inline bf16 f2bf(float f) {
  unsigned u = __builtin_bit_cast(unsigned, f);
  u += 0x7FFFu + ((u >> 16) & 1u);          // round to nearest even
  unsigned short s = (unsigned short)(u >> 16);
  return __builtin_bit_cast(bf16, s);
}

__device__ inline v8f wmma_bf16(v16bf a, v16bf b, v8f c) {
  // D(16x16,f32) = A(16x32,bf16) * B(32x16,bf16) + C
  return __builtin_amdgcn_wmma_f32_16x16x32_bf16(false, a, false, b,
                                                 (short)0, c, false, false);
}

// A fragment: 16x32 bf16 from bf16 row-major (row stride lda) -> two 16B loads.
// lane = hl*16 + m ; elements j<8 -> K = k0+8*hl+j ; j>=8 -> K = k0+16+8*hl+(j-8)
__device__ inline v16bf load_a_bf16(const bf16* A, int lda, int row, int k0, int hl) {
  const bf16* p = A + (size_t)row * lda + k0 + hl * 8;
  v16bf a;
#pragma unroll
  for (int j = 0; j < 8; ++j) a[j] = p[j];
#pragma unroll
  for (int j = 0; j < 8; ++j) a[j + 8] = p[16 + j];
  return a;
}

// B fragment: 32x16. lane = hl*16 + n (column n0+n); element j<8 -> K = k0+8*hl+j ; j>=8 -> K = k0+16+8*hl+(j-8)
__device__ inline v16bf load_b_bf16nn(const bf16* Bm, int ldb, int k0, int n0, int lane) {
  const bf16* p = Bm + (size_t)(k0 + (lane >> 4) * 8) * ldb + n0 + (lane & 15);
  v16bf b;
#pragma unroll
  for (int j = 0; j < 8; ++j) b[j] = p[(size_t)j * ldb];
#pragma unroll
  for (int j = 0; j < 8; ++j) b[j + 8] = p[(size_t)(16 + j) * ldb];
  return b;
}
__device__ inline v16bf load_b_bf16nt(const bf16* Bm, int ldb, int k0, int n0, int lane) {
  // B[k][n] = Bm[n*ldb + k]
  v16bf b;
  const bf16* p = Bm + (size_t)(n0 + (lane & 15)) * ldb + k0 + (lane >> 4) * 8;
#pragma unroll
  for (int j = 0; j < 8; ++j) b[j] = p[j];
#pragma unroll
  for (int j = 0; j < 8; ++j) b[j + 8] = p[16 + j];
  return b;
}

// C/D store: element r -> M = m0 + r + 8*(lane>=16), N = n0 + (lane&15)
// Writes f32 (Cf) and/or bf16 (Ch) copies.
__device__ inline void store_c2(float* Cf, bf16* Ch, int ldc, int m0, int n0, int lane,
                                const v8f* acc, const float* bias, int relu, float scale) {
  int hl = lane >> 4, n = lane & 15;
#pragma unroll
  for (int t = 0; t < 4; ++t) {
    int col = n0 + t * 16 + n;
    float bv = bias ? bias[col] : 0.f;
#pragma unroll
    for (int r = 0; r < 8; ++r) {
      int row = m0 + r + 8 * hl;
      float v = acc[t][r] * scale + bv;
      if (relu) v = fmaxf(v, 0.f);
      if (Cf) Cf[(size_t)row * ldc + col] = v;
      if (Ch) Ch[(size_t)row * ldc + col] = f2bf(v);
    }
  }
}

// ---------------- TDM: stage a 64(K) x 64(N) bf16 weight tile into LDS -------
// LDS layout after DMA: 64 rows x 66 bf16 (hardware pads 1 DWORD per 128B row
// via pad_interval=32dw/pad_amount=1dw -> odd word stride, conflict-free).
constexpr int BROW = 66;

__device__ inline void tdm_issue_b(const bf16* Wb, int ldb, int K,
                                   int k0, int n0, unsigned ldsOff) {
  unsigned long long ga =
      (unsigned long long)(const void*)(Wb + (size_t)k0 * ldb + n0);
  v4u g0;
  g0[0] = 1u;                                        // count=1, normal mode
  g0[1] = ldsOff;                                    // LDS byte address
  g0[2] = (unsigned)(ga & 0xFFFFFFFFu);              // global_addr[31:0]
  g0[3] = (unsigned)((ga >> 32) & 0x01FFFFFFu) | (2u << 30);  // [56:32] | type=2
  unsigned td0 = (unsigned)ldb, td1 = (unsigned)K;
  v8i g1;
  g1[0] = (int)0x01110000u;  // data_size=2B | pad_enable | pad_interval=32dw | pad_amount=1dw
  g1[1] = (int)((td0 & 0xFFFFu) << 16);                       // tensor_dim0 lo
  g1[2] = (int)((td0 >> 16) | ((td1 & 0xFFFFu) << 16));       // dim0 hi | dim1 lo
  g1[3] = (int)((td1 >> 16) | (64u << 16));                   // dim1 hi | tile_dim0=64
  g1[4] = (int)64u;                                           // tile_dim1=64
  g1[5] = (int)td0;                                           // tensor_dim0_stride
  g1[6] = 0; g1[7] = 0;
  v4i gz4 = {0, 0, 0, 0};
  v8i gz8 = {0, 0, 0, 0, 0, 0, 0, 0};
  // 6-arg form (this toolchain): (g0, g1, g2, g3, g4, cpol)
  __builtin_amdgcn_tensor_load_to_lds(g0, g1, gz4, gz4, gz8, 0);
}

// ---------------- block-cooperative GEMM: C = A(bf16)[M,K] * Wt(bf16)[K,N] ---
// block tile = 128(M) x 64(N); 8 waves each own a 16x64 slice and share the
// TDM-staged B tile; double-buffered over 64-deep K slices.
__global__ void __launch_bounds__(256)
gemm_xw(const bf16* A, long sAb, int lda,
        const bf16* W, long sWe, int ldb,
        const float* bias, long sBe,
        float* Cf, bf16* Ch, long sCb, int ldc,
        int M, int N, int K,
        const int* idx, const int* go, int relu) {
  __shared__ bf16 Bs[2][64][BROW];
  int b = blockIdx.z;
  if (go && !go[b]) return;                 // block-uniform
  int e = idx ? idx[b] : 0;
  const bf16*  Ab = A + (size_t)b * sAb;
  const bf16*  Wb = W + (size_t)e * sWe;
  const float* Bi = bias ? bias + (size_t)e * sBe : nullptr;
  float* Cfb = Cf ? Cf + (size_t)b * sCb : nullptr;
  bf16*  Chb = Ch ? Ch + (size_t)b * sCb : nullptr;

  int lane = threadIdx.x & 31, wave = threadIdx.x >> 5;
  int tilesN = N >> 6;
  int tM = blockIdx.x / tilesN, tN = blockIdx.x % tilesN;
  if (tM >= (M >> 7)) return;               // block-uniform (never with exact grid)
  int m0 = (tM << 7) + (wave << 4);
  int n0 = tN << 6;
  int hl = lane >> 4, mrow = m0 + (lane & 15);
  int nk = K >> 6;

  if (wave == 0) {
    tdm_issue_b(Wb, ldb, K, 0, n0, (unsigned)(unsigned long long)(void*)&Bs[0][0][0]);
    __builtin_amdgcn_s_wait_tensorcnt(0);
  }
  __syncthreads();

  v8f zero = {0.f, 0.f, 0.f, 0.f, 0.f, 0.f, 0.f, 0.f};
  v8f acc[4] = {zero, zero, zero, zero};
  for (int ks = 0; ks < nk; ++ks) {
    int buf = ks & 1;
    if (wave == 0 && ks + 1 < nk)
      tdm_issue_b(Wb, ldb, K, (ks + 1) << 6, n0,
                  (unsigned)(unsigned long long)(void*)&Bs[buf ^ 1][0][0]);
#pragma unroll
    for (int kc = 0; kc < 64; kc += 32) {
      v16bf a = load_a_bf16(Ab, lda, mrow, (ks << 6) + kc, hl);
#pragma unroll
      for (int t = 0; t < 4; ++t) {
        int nc = t * 16 + (lane & 15);
        v16bf bfr;
#pragma unroll
        for (int j = 0; j < 8; ++j) bfr[j] = Bs[buf][kc + 8 * hl + j][nc];
#pragma unroll
        for (int j = 0; j < 8; ++j) bfr[j + 8] = Bs[buf][kc + 16 + 8 * hl + j][nc];
        acc[t] = wmma_bf16(a, bfr, acc[t]);
      }
    }
    if (wave == 0 && ks + 1 < nk) __builtin_amdgcn_s_wait_tensorcnt(0);
    __syncthreads();
  }
  store_c2(Cfb, Chb, ldc, m0, n0, lane, acc, Bi, relu, 1.f);
}

// ---------------- attention: scores(f32) = Q * K^T * scale -------------------
__global__ void __launch_bounds__(256)
gemm_qk(const bf16* qkv, float* sc, const int* go, float scale) {
  int z = blockIdx.z, b = z >> 3, h = z & 7;
  if (!go[b]) return;
  const bf16* Q  = qkv + (size_t)b * SS * 3 * DD + h * DHH;
  const bf16* Kp = qkv + (size_t)b * SS * 3 * DD + DD + h * DHH;
  float* C = sc + (size_t)z * SS * SS;

  int lane = threadIdx.x & 31, wave = threadIdx.x >> 5;
  int wt = blockIdx.x * 8 + wave;
  int tilesN = SS >> 6;
  int tM = wt / tilesN, tN = wt % tilesN;
  if (tM >= (SS >> 4)) return;
  int m0 = tM << 4, n0 = tN << 6;
  int hl = lane >> 4, mrow = m0 + (lane & 15);

  v8f zero = {0.f, 0.f, 0.f, 0.f, 0.f, 0.f, 0.f, 0.f};
  v8f acc[4] = {zero, zero, zero, zero};
  for (int k0 = 0; k0 < DHH; k0 += 32) {
    v16bf a = load_a_bf16(Q, 3 * DD, mrow, k0, hl);
#pragma unroll
    for (int t = 0; t < 4; ++t) {
      v16bf bf = load_b_bf16nt(Kp, 3 * DD, k0, n0 + t * 16, lane);
      acc[t] = wmma_bf16(a, bf, acc[t]);
    }
  }
  store_c2(C, nullptr, SS, m0, n0, lane, acc, nullptr, 0, scale);
}

// ---------------- attention: ctx(bf16) = P(bf16) * V(bf16) -------------------
__global__ void __launch_bounds__(256)
gemm_pv(const bf16* pr, const bf16* qkv, bf16* ctx, const int* go) {
  int z = blockIdx.z, b = z >> 3, h = z & 7;
  if (!go[b]) return;
  const bf16* P = pr + (size_t)z * SS * SS;
  const bf16* V = qkv + (size_t)b * SS * 3 * DD + 2 * DD + h * DHH;
  bf16* C = ctx + (size_t)b * SS * DD + h * DHH;

  int lane = threadIdx.x & 31, wave = threadIdx.x >> 5;
  int wt = blockIdx.x * 8 + wave;          // tilesN = 1 (N = 64)
  if (wt >= (SS >> 4)) return;
  int m0 = wt << 4;
  int hl = lane >> 4, mrow = m0 + (lane & 15);

  v8f zero = {0.f, 0.f, 0.f, 0.f, 0.f, 0.f, 0.f, 0.f};
  v8f acc[4] = {zero, zero, zero, zero};
  for (int k0 = 0; k0 < SS; k0 += 32) {
    v16bf a = load_a_bf16(P, SS, mrow, k0, hl);
#pragma unroll
    for (int t = 0; t < 4; ++t) {
      v16bf bf = load_b_bf16nn(V, 3 * DD, k0, t * 16, lane);
      acc[t] = wmma_bf16(a, bf, acc[t]);
    }
  }
  store_c2(nullptr, C, DD, m0, 0, lane, acc, nullptr, 0, 1.f);
}

// -------- softmax rows of length S: f32 in -> bf16 probs out (wave/row) ------
__global__ void __launch_bounds__(256)
softmax_rows(const float* sc, bf16* pr, const int* go) {
  int row = blockIdx.x * 8 + (threadIdx.x >> 5);
  if (row >= BB * HH * SS) return;
  int b = row / (HH * SS);
  if (!go[b]) return;
  const float* p = sc + (size_t)row * SS;
  bf16* q = pr + (size_t)row * SS;
  int lane = threadIdx.x & 31;
  float v[8], mx = -1e30f;
#pragma unroll
  for (int j = 0; j < 8; ++j) { v[j] = p[lane + 32 * j]; mx = fmaxf(mx, v[j]); }
  for (int o = 16; o; o >>= 1) mx = fmaxf(mx, __shfl_xor(mx, o, 32));
  float s = 0.f;
#pragma unroll
  for (int j = 0; j < 8; ++j) { v[j] = __expf(v[j] - mx); s += v[j]; }
  for (int o = 16; o; o >>= 1) s += __shfl_xor(s, o, 32);
  float inv = 1.f / s;
#pragma unroll
  for (int j = 0; j < 8; ++j) q[lane + 32 * j] = f2bf(v[j] * inv);
}

// -------- LayerNorm(xa+xb)*g+b (+tag); writes f32 + optional bf16 mirror -----
__global__ void __launch_bounds__(256)
ln_kernel(const float* xa, const float* xb, const float* g, const float* beta,
          const float* tg, float* out, bf16* out16, const int* idx, const int* go) {
  int row = blockIdx.x * 8 + (threadIdx.x >> 5);
  if (row >= BB * SS) return;
  int b = row / SS;
  if (!go[b]) return;
  int e = idx[b];
  const float* pa = xa + (size_t)row * DD;
  const float* pb = xb + (size_t)row * DD;
  int lane = threadIdx.x & 31;
  float v[16], s = 0.f;
#pragma unroll
  for (int j = 0; j < 16; ++j) { int d = lane + 32 * j; v[j] = pa[d] + pb[d]; s += v[j]; }
  for (int o = 16; o; o >>= 1) s += __shfl_xor(s, o, 32);
  float mu = s * (1.f / DD), s2 = 0.f;
#pragma unroll
  for (int j = 0; j < 16; ++j) { float dq = v[j] - mu; s2 += dq * dq; }
  for (int o = 16; o; o >>= 1) s2 += __shfl_xor(s2, o, 32);
  float inv = rsqrtf(s2 * (1.f / DD) + 1e-5f);
#pragma unroll
  for (int j = 0; j < 16; ++j) {
    int d = lane + 32 * j;
    float o = (v[j] - mu) * inv * g[(size_t)e * DD + d] + beta[(size_t)e * DD + d];
    if (tg) o += tg[(size_t)e * DD + d];
    out[(size_t)row * DD + d] = o;
    if (out16) out16[(size_t)row * DD + d] = f2bf(o);
  }
}

// ---------------- router: mean-pool -> MLP -> mask/softmax/entropy/argmax ----
__global__ void __launch_bounds__(256)
router_kernel(const float* rep, const float* r_w1, const float* r_b1,
              const float* r_w2, const float* r_b2,
              int* visits, int* go, int* idxA, float* entAcc, int* activeA, int step) {
  __shared__ float sSum[BB][DD];
  __shared__ float sH[BB][RHH];
  __shared__ float sLg[BB][EP1];
  int t = threadIdx.x;
  if (step == 0) {
    if (t < BB * EE) visits[t] = 0;
    if (t < BB) activeA[t] = 1;
    if (t == 0) entAcc[0] = 0.f;
  }
  __threadfence_block();
  __syncthreads();
  for (int o = t; o < BB * DD; o += 256) {
    int b = o / DD, d = o % DD;
    const float* p = rep + (size_t)b * SS * DD + d;
    float s = 0.f;
    for (int si = 0; si < SS; ++si) s += p[(size_t)si * DD];
    sSum[b][d] = s * (1.f / SS);
  }
  __syncthreads();
  for (int o = t; o < BB * RHH; o += 256) {
    int b = o / RHH, r = o % RHH;
    float a = r_b1[r];
    const float* w = r_w1 + (size_t)r * DD;
    for (int d = 0; d < DD; ++d) a += sSum[b][d] * w[d];
    sH[b][r] = fmaxf(a, 0.f);
  }
  __syncthreads();
  for (int o = t; o < BB * EP1; o += 256) {
    int b = o / EP1, j = o % EP1;
    float a = r_b2[j];
    const float* w = r_w2 + (size_t)j * RHH;
    for (int r = 0; r < RHH; ++r) a += sH[b][r] * w[r];
    sLg[b][j] = a;
  }
  __syncthreads();
  if (t == 0) {
    int cnt = 0;
    for (int b = 0; b < BB; ++b) cnt += activeA[b];
    float entSum = 0.f;
    for (int b = 0; b < BB; ++b) {
      float lg[EP1];
      for (int j = 0; j < EE; ++j)
        lg[j] = (visits[b * EE + j] >= MAX_VISITS) ? NEGV : sLg[b][j];
      lg[EE] = sLg[b][EE];
      float mx = lg[0];
      for (int j = 1; j < EP1; ++j) mx = fmaxf(mx, lg[j]);
      float ps[EP1], ssum = 0.f;
      for (int j = 0; j < EP1; ++j) { ps[j] = __expf(lg[j] - mx); ssum += ps[j]; }
      float ent = 0.f;
      for (int j = 0; j < EP1; ++j) {
        float pj = ps[j] / ssum;
        ent -= pj * logf(pj + 1e-9f);
      }
      if (activeA[b]) entSum += ent;
      int choice = 0; float best = lg[0];
      for (int j = 1; j < EP1; ++j) if (lg[j] > best) { best = lg[j]; choice = j; }
      int g = (activeA[b] && choice < EE) ? 1 : 0;
      int e = g ? choice : 0;
      if (g) visits[b * EE + e] += 1;
      go[b] = g; idxA[b] = e;
    }
    if (cnt > 0) entAcc[0] += entSum / (float)cnt;
    for (int b = 0; b < BB; ++b) activeA[b] = go[b];
  }
}

// ---------------- embedding + positional encoding (f32 + bf16 mirror) --------
__global__ void __launch_bounds__(256)
embed_kernel(const int* ids, const float* emb, const float* pe,
             float* rep, bf16* rep16) {
  size_t i = (size_t)blockIdx.x * 256 + threadIdx.x;
  if (i >= (size_t)BB * SS * DD) return;
  int d = (int)(i % DD);
  size_t bs = i / DD;
  int s = (int)(bs % SS);
  int b = (int)(bs / SS);
  int id = ids[b * SS + s];
  float ev = (id != 0) ? emb[(size_t)id * DD + d] : 0.f;
  float r = ev * 22.62741699796952f /* sqrt(512) */ + pe[(size_t)s * DD + d];
  rep[i] = r;
  rep16[i] = f2bf(r);
}

// ---------------- weight prep: f32 [O,I] -> bf16 transposed [I,O] ------------
__global__ void __launch_bounds__(256)
wconv_kernel(const float* src, bf16* dst, int O, int I) {
  size_t z = blockIdx.z;
  src += z * (size_t)O * I;
  dst += z * (size_t)O * I;
  size_t i = (size_t)blockIdx.x * 256 + threadIdx.x;
  if (i >= (size_t)O * I) return;
  int o = (int)(i / I), c = (int)(i % I);
  dst[(size_t)c * O + o] = f2bf(src[i]);
}

__global__ void finalize_kernel(const float* ent, float* out) { out[0] = ent[0]; }

// ---------------- host launch -------------------------------------------------
extern "C" void kernel_launch(void* const* d_in, const int* in_sizes, int n_in,
                              void* d_out, int out_size, void* d_ws, size_t ws_size,
                              hipStream_t stream) {
  (void)in_sizes; (void)n_in; (void)out_size; (void)ws_size;
  const int*   input_ids = (const int*)  d_in[0];
  const float* emb   = (const float*)d_in[1];
  const float* pe    = (const float*)d_in[2];
  const float* Wqkv  = (const float*)d_in[3];
  const float* bqkv  = (const float*)d_in[4];
  const float* Wo    = (const float*)d_in[5];
  const float* bo    = (const float*)d_in[6];
  const float* W1    = (const float*)d_in[7];
  const float* b1    = (const float*)d_in[8];
  const float* W2    = (const float*)d_in[9];
  const float* b2    = (const float*)d_in[10];
  const float* ln1g  = (const float*)d_in[11];
  const float* ln1b  = (const float*)d_in[12];
  const float* ln2g  = (const float*)d_in[13];
  const float* ln2b  = (const float*)d_in[14];
  const float* tag   = (const float*)d_in[15];
  const float* r_w1  = (const float*)d_in[16];
  const float* r_b1  = (const float*)d_in[17];
  const float* r_w2  = (const float*)d_in[18];
  const float* r_b2  = (const float*)d_in[19];
  const float* lm_w  = (const float*)d_in[20];
  const float* lm_b  = (const float*)d_in[21];
  float* out = (float*)d_out;

  char* ws = (char*)d_ws;
  size_t off = 0;
  auto carve = [&](size_t bytes) { size_t o = off; off += (bytes + 255) & ~(size_t)255; return o; };
  float* repF   = (float*)(ws + carve((size_t)BB * SS * DD * 4));
  bf16*  repB   = (bf16*)(ws + carve((size_t)BB * SS * DD * 2));
  bf16*  qkvB   = (bf16*)(ws + carve((size_t)BB * SS * 3 * DD * 2));
  float* scF    = (float*)(ws + carve((size_t)BB * HH * SS * SS * 4));
  bf16*  prB    = (bf16*)(ws + carve((size_t)BB * HH * SS * SS * 2));
  bf16*  ctxB   = (bf16*)(ws + carve((size_t)BB * SS * DD * 2));
  float* tmpF   = (float*)(ws + carve((size_t)BB * SS * DD * 4));
  float* x1F    = (float*)(ws + carve((size_t)BB * SS * DD * 4));
  bf16*  x1B    = (bf16*)(ws + carve((size_t)BB * SS * DD * 2));
  bf16*  hidB   = (bf16*)(ws + carve((size_t)BB * SS * FFF * 2));
  bf16*  WqkvT  = (bf16*)(ws + carve((size_t)EE * DD * 3 * DD * 2));
  bf16*  WoT    = (bf16*)(ws + carve((size_t)EE * DD * DD * 2));
  bf16*  W1T    = (bf16*)(ws + carve((size_t)EE * DD * FFF * 2));
  bf16*  W2T    = (bf16*)(ws + carve((size_t)EE * FFF * DD * 2));
  bf16*  lmT    = (bf16*)(ws + carve((size_t)DD * VV * 2));
  int*   visits = (int*)(ws + carve(BB * EE * 4));
  int*   goA    = (int*)(ws + carve(BB * 4));
  int*   idxA   = (int*)(ws + carve(BB * 4));
  int*   actA   = (int*)(ws + carve(BB * 4));
  float* entA   = (float*)(ws + carve(4));

  auto blks = [](size_t n) { return (unsigned)((n + 255) / 256); };

  // --- weight prep (bf16 + transpose) ---
  wconv_kernel<<<dim3(blks((size_t)3 * DD * DD), 1, EE), 256, 0, stream>>>(Wqkv, WqkvT, 3 * DD, DD);
  wconv_kernel<<<dim3(blks((size_t)DD * DD), 1, EE), 256, 0, stream>>>(Wo, WoT, DD, DD);
  wconv_kernel<<<dim3(blks((size_t)FFF * DD), 1, EE), 256, 0, stream>>>(W1, W1T, FFF, DD);
  wconv_kernel<<<dim3(blks((size_t)DD * FFF), 1, EE), 256, 0, stream>>>(W2, W2T, DD, FFF);
  wconv_kernel<<<dim3(blks((size_t)VV * DD), 1, 1), 256, 0, stream>>>(lm_w, lmT, VV, DD);

  // --- embedding ---
  embed_kernel<<<blks((size_t)BB * SS * DD), 256, 0, stream>>>(input_ids, emb, pe, repF, repB);

  // --- routed expert steps ---
  for (int step = 0; step < MAX_PATH; ++step) {
    router_kernel<<<1, 256, 0, stream>>>(repF, r_w1, r_b1, r_w2, r_b2,
                                         visits, goA, idxA, entA, actA, step);
    // qkv(bf16) = rep @ Wqkv[e]^T + bqkv[e] : blocks = 2*24 = 48
    gemm_xw<<<dim3(48, 1, BB), 256, 0, stream>>>(
        repB, (long)SS * DD, DD, WqkvT, (long)DD * 3 * DD, 3 * DD,
        bqkv, (long)3 * DD, nullptr, qkvB, (long)SS * 3 * DD, 3 * DD,
        SS, 3 * DD, DD, idxA, goA, 0);
    gemm_qk<<<dim3(8, 1, BB * HH), 256, 0, stream>>>(qkvB, scF, goA, 0.125f);
    softmax_rows<<<dim3((BB * HH * SS) / 8), 256, 0, stream>>>(scF, prB, goA);
    gemm_pv<<<dim3(2, 1, BB * HH), 256, 0, stream>>>(prB, qkvB, ctxB, goA);
    // attn_out(f32) = ctx @ Wo[e]^T + bo[e] : blocks = 2*8 = 16
    gemm_xw<<<dim3(16, 1, BB), 256, 0, stream>>>(
        ctxB, (long)SS * DD, DD, WoT, (long)DD * DD, DD,
        bo, (long)DD, tmpF, nullptr, (long)SS * DD, DD,
        SS, DD, DD, idxA, goA, 0);
    // x1 = LN(rep + attn_out)  (f32 + bf16 mirror)
    ln_kernel<<<dim3((BB * SS) / 8), 256, 0, stream>>>(repF, tmpF, ln1g, ln1b,
                                                       nullptr, x1F, x1B, idxA, goA);
    // hid(bf16) = relu(x1 @ W1[e]^T + b1[e]) : blocks = 2*32 = 64
    gemm_xw<<<dim3(64, 1, BB), 256, 0, stream>>>(
        x1B, (long)SS * DD, DD, W1T, (long)DD * FFF, FFF,
        b1, (long)FFF, nullptr, hidB, (long)SS * FFF, FFF,
        SS, FFF, DD, idxA, goA, 1);
    // ff(f32) = hid @ W2[e]^T + b2[e]       : blocks = 2*8 = 16
    gemm_xw<<<dim3(16, 1, BB), 256, 0, stream>>>(
        hidB, (long)SS * FFF, FFF, W2T, (long)FFF * DD, DD,
        b2, (long)DD, tmpF, nullptr, (long)SS * DD, DD,
        SS, DD, FFF, idxA, goA, 0);
    // rep = LN(x1 + ff) + tag[e]   (gated write, f32 + bf16 mirror)
    ln_kernel<<<dim3((BB * SS) / 8), 256, 0, stream>>>(x1F, tmpF, ln2g, ln2b,
                                                       tag, repF, repB, idxA, goA);
  }

  // --- LM head: [B*S, D] x [D, V] + lm_b -> d_out : blocks = 16*500 = 8000 ---
  {
    int M = BB * SS;                       // 2048
    gemm_xw<<<dim3((M / 128) * (VV / 64), 1, 1), 256, 0, stream>>>(
        repB, 0L, DD, lmT, 0L, VV,
        lm_b, 0L, out, nullptr, 0L, VV,
        M, VV, DD, nullptr, nullptr, 0);
  }
  finalize_kernel<<<1, 1, 0, stream>>>(entA, out + (size_t)BB * SS * VV);
}
